// LstmSpeakerEncoder_13632226197636
// MI455X (gfx1250) — matrix operand order repaired and checked by reading, not compiled, not run on hardware
//
#include <hip/hip_runtime.h>
#include <cstdint>

#define NFILT 40
#define NBINS 257
#define HID   64
#define BB    64
#define TT    2000
#define MROWS (BB*TT)          // 128000
#define GATES 256              // 4*HID

typedef __attribute__((ext_vector_type(16))) __bf16 v16bf;
typedef __attribute__((ext_vector_type(8)))  float  v8f;

union BF16Frag { v16bf v; uint32_t u[8]; uint4 q[2]; };
union H8 { uint4 q; _Float16 h[8]; };

__device__ __forceinline__ unsigned short f2bf(float f) {
    union { __bf16 h; unsigned short u; } x;
    x.h = (__bf16)f;                                  // native RNE convert
    return x.u;
}
__device__ __forceinline__ float bf2f(unsigned short v) {
    union { uint32_t u; float f; } x; x.u = ((uint32_t)v) << 16;
    return x.f;
}
__device__ __forceinline__ uint32_t pack2(float a, float b) {
    union { __bf16 h[2]; uint32_t u; } x;
    x.h[0] = (__bf16)a; x.h[1] = (__bf16)b;
    return x.u;
}
__device__ __forceinline__ float sigm(float x) { return 1.0f / (1.0f + __expf(-x)); }

// ---------------------------------------------------------------- fbank ----
__device__ float fbval(const float* b, int f, int n) {
    if (f >= NFILT - 1 || n >= NBINS) return 0.0f;   // last filter masked + pads
    float bj = b[f], bj1 = b[f + 1], bj2 = b[f + 2];
    float fb0 = floorf(bj), fb1 = floorf(bj1), fb2 = floorf(bj2);
    float fn = (float)n;
    bool rise = (fn >= fb0) && (fn < fb1);
    bool fall = (fn >= fb1) && (fn < fb2);
    float d1 = bj1 - bj, d2 = bj2 - bj1;
    float dd1 = d1 > 0.f ? d1 : 1.f, dd2 = d2 > 0.f ? d2 : 1.f;
    float rv = (fn - bj) / (dd1 * dd1);
    float fv = (bj2 - fn) / (dd2 * dd2);
    return fall ? fv : (rise ? rv : 0.0f);
}

// fbT fragment-linear B operand: [ck(9)][nt(3)][lane(32)][j(8)] uint32 (bf16 k-pairs)
__global__ void fbank_kernel(const float* __restrict__ binpoints, uint32_t* __restrict__ fbT) {
    __shared__ float b[NFILT + 2];
    if (threadIdx.x == 0) {
        for (int i = 0; i < NFILT + 2; ++i) b[i] = binpoints[i];
        for (int i = 1; i < NFILT + 2; ++i) {       // insertion sort (42 elems)
            float v = b[i]; int j = i - 1;
            while (j >= 0 && b[j] > v) { b[j + 1] = b[j]; --j; }
            b[j + 1] = v;
        }
    }
    __syncthreads();
    for (int i = threadIdx.x; i < 9 * 3 * 32 * 8; i += blockDim.x) {
        int ck = i / 768, r = i % 768;
        int nt = r >> 8, r2 = r & 255;
        int lane = r2 >> 3, j = r2 & 7;
        int f = nt * 16 + (lane & 15);
        int k = ck * 32 + (j >> 2) * 16 + (lane >> 4) * 8 + 2 * (j & 3);
        fbT[i] = pack2(fbval(b, f, k), fbval(b, f, k + 1));
    }
}

__global__ void zero_u32(uint32_t* p, unsigned n) {
    for (unsigned i = blockIdx.x * blockDim.x + threadIdx.x; i < n; i += gridDim.x * blockDim.x)
        p[i] = 0u;
}

// ------------------------------------------------- filterbank GEMM + log ---
// out0: bf16 [MROWS][64] (cols 40..63 pre-zeroed)
__global__ void __launch_bounds__(256) filt_gemm(const float* __restrict__ x,
                                                 const uint32_t* __restrict__ fbT,
                                                 unsigned short* __restrict__ out0) {
    __shared__ uint32_t lf[9 * 3 * 32 * 8];          // 27648 B
    __shared__ unsigned short lx[128 * 32];          // 8 KB staged A chunk
    for (int i = threadIdx.x; i < 9 * 3 * 32 * 8; i += 256) lf[i] = fbT[i];
    int wave = threadIdx.x >> 5, lane = threadIdx.x & 31;
    int hi = lane >> 4, lm = lane & 15;
    int rowbase = blockIdx.x * 128;
    v8f acc[3] = {};
    for (int ck = 0; ck < 9; ++ck) {
        __syncthreads();                             // lx reuse + (ck==0) lf ready
        if (ck < 8) {                                // full chunk: unconditional burst
#pragma unroll
            for (int u = 0; u < 16; ++u) {
                int i = threadIdx.x + u * 256;
                int row = i >> 5, c = i & 31;
                lx[i] = f2bf(x[(size_t)(rowbase + row) * NBINS + ck * 32 + c]);
            }
        } else {                                     // tail: only k==256 is real
#pragma unroll
            for (int u = 0; u < 16; ++u) {
                int i = threadIdx.x + u * 256;
                int row = i >> 5, c = i & 31;
                unsigned short v = 0;
                if (c == 0) v = f2bf(x[(size_t)(rowbase + row) * NBINS + 256]);
                lx[i] = v;
            }
        }
        __syncthreads();
        const char* ap = (const char*)lx + (wave * 16 + lm) * 64;
        BF16Frag a;
        a.q[0] = *(const uint4*)(ap + hi * 16);
        a.q[1] = *(const uint4*)(ap + 32 + hi * 16);
#pragma unroll
        for (int nt = 0; nt < 3; ++nt) {
            BF16Frag bb;
            int base = ((ck * 3 + nt) * 32 + lane) * 8;
            bb.q[0] = *(const uint4*)&lf[base];
            bb.q[1] = *(const uint4*)&lf[base + 4];
            acc[nt] = __builtin_amdgcn_wmma_f32_16x16x32_bf16(
                false, a.v, false, bb.v, (short)0, acc[nt], false, false);
        }
    }
#pragma unroll
    for (int nt = 0; nt < 3; ++nt) {
        int f = nt * 16 + lm;
        if (f >= NFILT) continue;
#pragma unroll
        for (int r = 0; r < 8; ++r) {
            int m = rowbase + wave * 16 + hi * 8 + r;
            float v = acc[nt][r];
            if (f == 0) v = x[(size_t)m * NBINS];    // filt[:,:,0] = x[:,:,0]
            out0[(size_t)m * 64 + f] = f2bf(__logf(v + 1e-10f));
        }
    }
}

// ------------------------------------------- per-layer weight repacking ----
// wihT frag-linear: [d][ck(Kpad/32)][nt(16)][lane][j]; whhT: [d][ck(2)][nt(16)][lane][j]
__global__ void conv_weights(const float* __restrict__ wih, const float* __restrict__ whh,
                             const float* __restrict__ bih, const float* __restrict__ bhh,
                             int Kreal, int Kpad,
                             uint32_t* __restrict__ wihT, uint32_t* __restrict__ whhT,
                             float* __restrict__ bsum) {
    int stride = gridDim.x * blockDim.x;
    int CK = Kpad >> 5;
    int per_d = CK * 4096;
    for (int i = blockIdx.x * blockDim.x + threadIdx.x; i < 2 * per_d; i += stride) {
        int d = i / per_d, r = i % per_d;
        int ck = r / 4096, r2 = r & 4095;
        int nt = r2 >> 8, lane = (r2 >> 3) & 31, j = r2 & 7;
        int g = nt * 16 + (lane & 15);
        int k = ck * 32 + (j >> 2) * 16 + (lane >> 4) * 8 + 2 * (j & 3);
        float lo = (k < Kreal) ? wih[((size_t)d * GATES + g) * Kreal + k] : 0.f;
        float hic = (k + 1 < Kreal) ? wih[((size_t)d * GATES + g) * Kreal + k + 1] : 0.f;
        wihT[i] = pack2(lo, hic);
    }
    for (int i = blockIdx.x * blockDim.x + threadIdx.x; i < 2 * 8192; i += stride) {
        int d = i / 8192, r = i % 8192;
        int ck = r >> 12, r2 = r & 4095;
        int nt = r2 >> 8, lane = (r2 >> 3) & 31, j = r2 & 7;
        int g = nt * 16 + (lane & 15);
        int k = ck * 32 + (j >> 2) * 16 + (lane >> 4) * 8 + 2 * (j & 3);
        float lo = whh[((size_t)d * GATES + g) * HID + k];
        float hic = whh[((size_t)d * GATES + g) * HID + k + 1];
        whhT[i] = pack2(lo, hic);
    }
    for (int i = blockIdx.x * blockDim.x + threadIdx.x; i < 2 * 256; i += stride)
        bsum[i] = bih[i] + bhh[i];
}

// ----------------------------------- input projection: pre = h*Wih^T + b ---
// grid (1000, 4, 2); block 256 (8 waves, 1 Mtile/wave, 4 Ntiles/wave)
// pre layout: [d][t][b][g]  (scan touches one contiguous 64KB block per step)
template <int CK>
__global__ void __launch_bounds__(256) pre_gemm(const unsigned short* __restrict__ inH,
                                                const uint32_t* __restrict__ wihT,
                                                const float* __restrict__ bsum,
                                                float* __restrict__ pre) {
    constexpr int Kp = CK * 32;
    int d = blockIdx.z;
    __shared__ uint32_t lw[CK * 4 * 32 * 8];         // [ck][ntl(4)][lane][j]
#pragma unroll
    for (int u = 0; u < CK * 4; ++u) {
        int i = threadIdx.x + u * 256;
        int ck = i >> 10, r = i & 1023;
        int ntl = r >> 8, rest = r & 255;
        lw[i] = wihT[(size_t)d * (CK * 4096) + ck * 4096 + (blockIdx.y * 4 + ntl) * 256 + rest];
    }
    __syncthreads();
    int wave = threadIdx.x >> 5, lane = threadIdx.x & 31;
    int hi = lane >> 4, lm = lane & 15;
    int m0 = (blockIdx.x * 8 + wave) * 16;
    int mrow = m0 + lm;
    const uint4* A4 = (const uint4*)inH;
    v8f acc[4] = {};
#pragma unroll
    for (int ck = 0; ck < CK; ++ck) {
        BF16Frag a;
        size_t bidx = ((size_t)mrow * Kp + ck * 32 + hi * 8) >> 3;
        a.q[0] = A4[bidx];
        a.q[1] = A4[bidx + 2];
#pragma unroll
        for (int lt = 0; lt < 4; ++lt) {
            BF16Frag bb;
            int base = ((ck * 4 + lt) * 32 + lane) * 8;
            bb.q[0] = *(const uint4*)&lw[base];
            bb.q[1] = *(const uint4*)&lw[base + 4];
            acc[lt] = __builtin_amdgcn_wmma_f32_16x16x32_bf16(
                false, a.v, false, bb.v, (short)0, acc[lt], false, false);
        }
    }
#pragma unroll
    for (int lt = 0; lt < 4; ++lt) {
        int g = blockIdx.y * 64 + lt * 16 + lm;
        float bs = bsum[d * 256 + g];
#pragma unroll
        for (int r = 0; r < 8; ++r) {
            int m = m0 + hi * 8 + r;
            int b = m / TT, t = m - b * TT;
            pre[(((size_t)d * TT + t) * BB + b) * 256 + g] = acc[lt][r] + bs;
        }
    }
}

// ------------------------------------------- persistent recurrent scan ----
// grid 2 (one block per direction), block 256 (8 waves); Whh B-frags live in VGPRs.
// Split barrier after the GEMM phase: signal, overlap gate-phase pre loads, wait.
__global__ void __launch_bounds__(256, 1) lstm_scan(const float* __restrict__ pre,
                                                    const uint32_t* __restrict__ whhT,
                                                    unsigned short* __restrict__ outbuf) {
    int d = blockIdx.x;
    __shared__ unsigned short lh[BB * HID];          // h (bf16), A operand (8 KB)
    __shared__ _Float16       lg[BB * GATES];        // h*Whh^T pre-acts (32 KB)
    const float* preD = pre + (size_t)d * TT * BB * 256;
    for (int i = threadIdx.x; i < BB * HID; i += 256) lh[i] = 0;
    int tid = threadIdx.x;
    int wave = tid >> 5, lane = tid & 31, hi = lane >> 4, lm = lane & 15;
    int mt = wave >> 1;                              // one M-tile per wave
    int rb = mt * 16;
    int cb = tid >> 2, jb = (tid & 3) * 16;          // fixed (batch, gate-quarter) owner
    // ---- hoist Whh B-fragments into registers (loop-invariant over 2000 steps) ----
    const uint4* W4 = (const uint4*)(whhT + (size_t)d * 8192);
    BF16Frag bfr[8][2];
#pragma unroll
    for (int i = 0; i < 8; ++i) {
        int nt = (wave & 1) * 8 + i;
        int base0 = (nt * 32 + lane) * 2;            // uint4 units
        int base1 = ((16 + nt) * 32 + lane) * 2;
        bfr[i][0].q[0] = W4[base0];
        bfr[i][0].q[1] = W4[base0 + 1];
        bfr[i][1].q[0] = W4[base1];
        bfr[i][1].q[1] = W4[base1 + 1];
    }
    float creg[16];
#pragma unroll
    for (int i = 0; i < 16; ++i) creg[i] = 0.0f;
    const uint4* lgq = (const uint4*)lg;
    __syncthreads();

    for (int t = 0; t < TT; ++t) {
        int tt = d ? (TT - 1 - t) : t;
        // ---- GEMM phase: lg = h * Whh^T (A frags hoisted: one mt per wave) ----
        const char* ap = (const char*)lh + (rb + lm) * (HID * 2);
        BF16Frag a0, a1;
        a0.q[0] = *(const uint4*)(ap + hi * 16);
        a0.q[1] = *(const uint4*)(ap + 32 + hi * 16);
        a1.q[0] = *(const uint4*)(ap + 64 + hi * 16);
        a1.q[1] = *(const uint4*)(ap + 96 + hi * 16);
#pragma unroll
        for (int i = 0; i < 8; ++i) {
            int nt = (wave & 1) * 8 + i;
            int gcol = nt * 16 + lm;
            v8f c8 = {};
            c8 = __builtin_amdgcn_wmma_f32_16x16x32_bf16(
                false, a0.v, false, bfr[i][0].v, (short)0, c8, false, false);
            c8 = __builtin_amdgcn_wmma_f32_16x16x32_bf16(
                false, a1.v, false, bfr[i][1].v, (short)0, c8, false, false);
#pragma unroll
            for (int r = 0; r < 8; ++r)
                lg[(rb + hi * 8 + r) * 256 + gcol] = (_Float16)c8[r];
        }
        // ---- split barrier: signal, overlap pre loads with stragglers, wait ----
        asm volatile("s_wait_dscnt 0x0" ::: "memory");
        asm volatile("s_barrier_signal -1" ::: "memory");
        const float* prow = preD + (size_t)tt * (BB * 256) + cb * 256;
        float pf[32];                                 // gate inputs for j = jb..jb+7
#pragma unroll
        for (int gq = 0; gq < 4; ++gq)
#pragma unroll
            for (int s = 0; s < 8; ++s)
                pf[gq * 8 + s] = prow[gq * 64 + jb + s];
        asm volatile("s_barrier_wait -1" ::: "memory");
        // ---- gate phase: g = lg + pre ; c in registers; h -> LDS + global ----
        uint32_t hpk[8];
#pragma unroll
        for (int half = 0; half < 2; ++half) {
            H8 ig8, fg8, gg8, og8;
            int off = (cb * 256 + jb + half * 8) >> 3;
            ig8.q = lgq[off];
            fg8.q = lgq[off + 8];
            gg8.q = lgq[off + 16];
            og8.q = lgq[off + 24];
#pragma unroll
            for (int s = 0; s < 8; s += 2) {
                float hp[2];
#pragma unroll
                for (int e = 0; e < 2; ++e) {
                    int jj = half * 8 + s + e;
                    int j = jb + jj;
                    float ig, fg, gg, og;
                    if (half == 0) {
                        ig = (float)ig8.h[s + e] + pf[s + e];
                        fg = (float)fg8.h[s + e] + pf[8 + s + e];
                        gg = (float)gg8.h[s + e] + pf[16 + s + e];
                        og = (float)og8.h[s + e] + pf[24 + s + e];
                    } else {
                        ig = (float)ig8.h[s + e] + prow[j];
                        fg = (float)fg8.h[s + e] + prow[64 + j];
                        gg = (float)gg8.h[s + e] + prow[128 + j];
                        og = (float)og8.h[s + e] + prow[192 + j];
                    }
                    float c = sigm(fg) * creg[jj] + sigm(ig) * tanhf(gg);
                    creg[jj] = c;
                    hp[e] = sigm(og) * tanhf(c);
                }
                hpk[(half * 8 + s) >> 1] = pack2(hp[0], hp[1]);
            }
        }
        uint4* lhv = (uint4*)((char*)lh + cb * (HID * 2) + jb * 2);
        uint4* orow = (uint4*)(outbuf + ((size_t)cb * TT + tt) * 128 + d * HID + jb);
        const uint4* hv = (const uint4*)hpk;
        lhv[0] = hv[0]; lhv[1] = hv[1];
        orow[0] = hv[0]; orow[1] = hv[1];
        if (t + 1 < TT) {                            // prefetch next step's pre block
            int tn = d ? (TT - 2 - t) : (t + 1);
            __builtin_prefetch(preD + (size_t)tn * (BB * 256) + tid * 64, 0, 0);
        }
        __syncthreads();
    }
}

// ------------------------------------------------------------- time mean ---
__global__ void mean_kernel(const unsigned short* __restrict__ buf, float* __restrict__ out) {
    int b = blockIdx.x, j = threadIdx.x;
    float s = 0.0f;
    for (int t = 0; t < TT; ++t)
        s += bf2f(buf[((size_t)b * TT + t) * 128 + j]);
    out[b * 128 + j] = s * (1.0f / (float)TT);
}

// ---------------------------------------------------------------- launch ---
extern "C" void kernel_launch(void* const* d_in, const int* in_sizes, int n_in,
                              void* d_out, int out_size, void* d_ws, size_t ws_size,
                              hipStream_t stream) {
    (void)in_sizes; (void)n_in; (void)out_size; (void)ws_size;
    const float* x         = (const float*)d_in[0];
    const float* binpoints = (const float*)d_in[1];

    char* ws = (char*)d_ws;
    size_t off = 0;
    auto alloc = [&](size_t bytes) -> void* {
        off = (off + 255) & ~(size_t)255;
        void* p = ws + off;
        off += bytes;
        return p;
    };
    uint32_t*       fbT  = (uint32_t*)      alloc((size_t)9 * 3 * 32 * 8 * 4);
    unsigned short* buf0 = (unsigned short*)alloc((size_t)MROWS * 64 * 2);
    unsigned short* bufA = (unsigned short*)alloc((size_t)MROWS * 128 * 2);
    unsigned short* bufB = (unsigned short*)alloc((size_t)MROWS * 128 * 2);
    float*          preB = (float*)         alloc((size_t)2 * MROWS * 256 * 4);
    uint32_t*       wihT = (uint32_t*)      alloc((size_t)2 * 4 * 4096 * 4);
    uint32_t*       whhT = (uint32_t*)      alloc((size_t)2 * 8192 * 4);
    float*          bsum = (float*)         alloc((size_t)2 * 256 * 4);

    fbank_kernel<<<1, 256, 0, stream>>>(binpoints, fbT);
    zero_u32<<<2048, 256, 0, stream>>>((uint32_t*)buf0, (unsigned)(MROWS * 64 / 2));
    filt_gemm<<<MROWS / 128, 256, 0, stream>>>(x, fbT, buf0);

    const unsigned short* in = buf0;
    unsigned short* outs[3] = {bufA, bufB, bufA};
    for (int l = 0; l < 3; ++l) {
        conv_weights<<<128, 256, 0, stream>>>(
            (const float*)d_in[2 + 4 * l], (const float*)d_in[3 + 4 * l],
            (const float*)d_in[4 + 4 * l], (const float*)d_in[5 + 4 * l],
            (l == 0) ? NFILT : 128, (l == 0) ? 64 : 128, wihT, whhT, bsum);
        if (l == 0)
            pre_gemm<2><<<dim3(MROWS / 128, 4, 2), 256, 0, stream>>>(in, wihT, bsum, preB);
        else
            pre_gemm<4><<<dim3(MROWS / 128, 4, 2), 256, 0, stream>>>(in, wihT, bsum, preB);
        lstm_scan<<<2, 256, 0, stream>>>(preB, whhT, outs[l]);
        in = outs[l];
    }
    mean_kernel<<<BB, 128, 0, stream>>>(bufA, (float*)d_out);
}
